// LinearAttention_37460704756492
// MI455X (gfx1250) — compile-verified, matrix operand order checked by reading, not run on hardware
//
#include <hip/hip_runtime.h>
#include <hip/hip_bf16.h>

#define B_   2
#define T_   2048
#define D_   512
#define H_   8
#define D2_  1024          // 2*D
#define MTOT_ (B_*T_)      // 4096 rows for the token-parallel GEMMs

typedef __attribute__((ext_vector_type(16))) __bf16 v16bf;
typedef __attribute__((ext_vector_type(8)))  __bf16 v8bf;
typedef __attribute__((ext_vector_type(8)))  float  v8f;

// ---------------------------------------------------------------------------
// Fragment load for the 16x16x32 bf16 WMMA.
// A-matrix 16x32 (MxK) per ISA: lane L (L&15 = row), holds K-segments
// [h*8, h*8+8) and [16+h*8, 16+h*8+8) with h = L>>4, packed sequentially
// into the 16-element fragment. The dense B operand (KxN) uses the mirrored
// layout with lane = N column, so a matrix stored N-major/K-contiguous
// (i.e. Bt row-major) loads with the *same* pattern.
// ---------------------------------------------------------------------------
__device__ __forceinline__ v16bf ld_frag(const __bf16* __restrict__ p, int ld,
                                         int row0, int k0) {
  const int lane = threadIdx.x & 31;
  const __bf16* q = p + (size_t)(row0 + (lane & 15)) * (size_t)ld
                      + (size_t)(k0 + ((lane >> 4) << 3));
  v8bf lo = *(const v8bf*)q;        // K = k0 + h*8 ..  +8
  v8bf hi = *(const v8bf*)(q + 16); // K = k0 + 16 + h*8 .. +8
  return __builtin_shufflevector(lo, hi, 0,1,2,3,4,5,6,7,8,9,10,11,12,13,14,15);
}

__device__ __forceinline__ v8f wmma_bf16(v16bf a, v16bf b, v8f c) {
  return __builtin_amdgcn_wmma_f32_16x16x32_bf16(false, a, false, b,
                                                 (short)0, c, false, false);
}

__device__ __forceinline__ v8f vzero8() {
  v8f z = {0.f,0.f,0.f,0.f,0.f,0.f,0.f,0.f};
  return z;
}

// ---------------------------------------------------------------------------
// K1: f32 -> bf16 elementwise convert (inputs and weights).
// ---------------------------------------------------------------------------
__global__ void k_cvt_bf16(const float* __restrict__ in, __bf16* __restrict__ out,
                           int n) {
  int i = blockIdx.x * 256 + threadIdx.x;
  if (i < n) out[i] = (__bf16)in[i];
}

// ---------------------------------------------------------------------------
// K2: per-head projections. QZ[b,h,t,e] = sum_d x[b,t,d]*Q[h,e,d]  (row-major)
//     PZt[b,h,e,t] = (1/H) * sum_d x[b,t,d]*P[h,e,d]               (transposed)
// QZ gets the 1/sqrt(D) score scale folded in. Q/P stored (e,d) are already
// the N-major/K-contiguous layout the B fragment wants.
// Each wave: one 16-row t-tile x 64-col e-strip, A fragment reused 4x (x2 mats).
// ---------------------------------------------------------------------------
__global__ void k_proj(const __bf16* __restrict__ xb, const __bf16* __restrict__ Qb,
                       const __bf16* __restrict__ Pb, __bf16* __restrict__ QZ,
                       __bf16* __restrict__ PZt) {
  int wt = blockIdx.x * 8 + (threadIdx.x >> 5);
  int et = wt & 7;                // D/64  = 8
  int tt = (wt >> 3) & 127;       // T/16  = 128
  int h  = (wt >> 10) & 7;        // H     = 8
  int b  = wt >> 13;              // B     = 2
  int t0 = tt * 16, e0 = et * 64;
  const __bf16* xA = xb + (size_t)b * T_ * D_;
  const __bf16* Qh = Qb + (size_t)h * D_ * D_;
  const __bf16* Ph = Pb + (size_t)h * D_ * D_;

  v8f cq[4], cp[4];
  #pragma unroll
  for (int j = 0; j < 4; ++j) { cq[j] = vzero8(); cp[j] = vzero8(); }

  for (int k = 0; k < D_; k += 32) {
    v16bf a = ld_frag(xA, D_, t0, k);
    #pragma unroll
    for (int j = 0; j < 4; ++j) {
      v16bf bq = ld_frag(Qh, D_, e0 + 16*j, k);
      cq[j] = wmma_bf16(a, bq, cq[j]);
      v16bf bp = ld_frag(Ph, D_, e0 + 16*j, k);
      cp[j] = wmma_bf16(a, bp, cp[j]);
    }
  }

  const float rs = 0.04419417382415922f;  // 1/sqrt(512) -> folded into QZ
  const float rh = 0.125f;                // 1/H         -> folded into PZ
  __bf16* QZh = QZ  + (size_t)(b*H_ + h) * T_ * D_;
  __bf16* PZh = PZt + (size_t)(b*H_ + h) * D_ * T_;
  const int lane = threadIdx.x & 31;
  const int n = lane & 15, m0 = (lane >> 4) << 3;
  #pragma unroll
  for (int j = 0; j < 4; ++j) {
    int e = e0 + 16*j + n;
    #pragma unroll
    for (int v = 0; v < 8; ++v)       // row-major QZ (strided b16 stores)
      QZh[(size_t)(t0 + m0 + v) * D_ + e] = (__bf16)(cq[j][v] * rs);
    v8bf pk;                           // transposed PZ: 8 contiguous, 16B store
    #pragma unroll
    for (int v = 0; v < 8; ++v) pk[v] = (__bf16)(cp[j][v] * rh);
    *(v8bf*)(PZh + (size_t)e * T_ + t0 + m0) = pk;
  }
}

// ---------------------------------------------------------------------------
// K3: fused causal linear attention for one (b, h, 64-row t-tile).
// 8 waves: wave = (e-half[0..1], m-band[0..3]); each wave owns a 16(t)x256(e)
// f32 accumulator (16 C tiles = 128 VGPRs).
// Per 32-wide s-step:
//   St[s,t] = sum_e x[s,e]*QZ[t,e]  (two 16x16 WMMAs, K-loop over D)
//   -> St's C layout IS the A-fragment layout for the next WMMA, so the
//      causal mask (s<=t) + f32->bf16 repack is purely in-lane.
//   y[t,e] += St-as-A  x  PZt-as-B  (16 WMMAs over the wave's e-strip)
// Head reduction via relaxed agent-scope f32 atomics into y.
// ---------------------------------------------------------------------------
__global__ void k_attn(const __bf16* __restrict__ xb, const __bf16* __restrict__ QZ,
                       const __bf16* __restrict__ PZt, float* __restrict__ y) {
  const int rt = blockIdx.x & 31;          // T/64 = 32 row tiles
  const int h  = (blockIdx.x >> 5) & 7;
  const int b  = blockIdx.x >> 8;
  const int w  = threadIdx.x >> 5;
  const int t0 = rt * 64 + (w & 3) * 16;   // wave's 16 t-rows
  const int e0 = (w >> 2) * 256;           // wave's 256-col e-half

  const __bf16* xs  = xb  + (size_t)b * T_ * D_;
  const __bf16* QZh = QZ  + (size_t)(b*H_ + h) * T_ * D_;
  const __bf16* PZh = PZt + (size_t)(b*H_ + h) * D_ * T_;

  v8f acc[16];
  #pragma unroll
  for (int j = 0; j < 16; ++j) acc[j] = vzero8();

  const int lane = threadIdx.x & 31;
  const int tcol = t0 + (lane & 15);
  const int s_end = t0 + 16;               // causal: only s <= t contribute

  for (int s0 = 0; s0 < s_end; s0 += 32) {
    v8f st0 = vzero8(), st1 = vzero8();
    for (int k = 0; k < D_; k += 32) {
      v16bf bq  = ld_frag(QZh, D_, t0, k);       // B: N = t
      v16bf ax0 = ld_frag(xs,  D_, s0, k);       // A: M = s (first 16)
      v16bf ax1 = ld_frag(xs,  D_, s0 + 16, k);  // A: M = s (second 16)
      st0 = wmma_bf16(ax0, bq, st0);
      st1 = wmma_bf16(ax1, bq, st1);
    }
    // mask + repack: St C-tile (lane = t col, VGPR = s row) -> A fragment
    const int srow = s0 + ((lane >> 4) << 3);
    v16bf aS;
    #pragma unroll
    for (int v = 0; v < 8; ++v) {
      float f0 = (srow + v      <= tcol) ? st0[v] : 0.0f;
      float f1 = (srow + 16 + v <= tcol) ? st1[v] : 0.0f;
      aS[v]     = (__bf16)f0;
      aS[8 + v] = (__bf16)f1;
    }
    #pragma unroll
    for (int j = 0; j < 16; ++j) {
      v16bf bp = ld_frag(PZh, T_, e0 + 16*j, s0);  // B: N = e, K = s
      acc[j] = wmma_bf16(aS, bp, acc[j]);
    }
  }

  const int m0 = (lane >> 4) << 3;
  #pragma unroll
  for (int j = 0; j < 16; ++j) {
    int e = e0 + 16*j + (lane & 15);
    #pragma unroll
    for (int v = 0; v < 8; ++v) {
      float* dst = y + ((size_t)b * T_ + (size_t)(t0 + m0 + v)) * D_ + e;
      __hip_atomic_fetch_add(dst, acc[j][v], __ATOMIC_RELAXED,
                             __HIP_MEMORY_SCOPE_AGENT);
    }
  }
}

// ---------------------------------------------------------------------------
// K4: z = LayerNorm(y + x) * gamma + beta ; emit f32 (residual) and bf16 copy.
// One block per token row; 256 threads x 2 elements.
// ---------------------------------------------------------------------------
__global__ void k_ln(const float* __restrict__ y, const float* __restrict__ x,
                     const float* __restrict__ gamma, const float* __restrict__ beta,
                     float* __restrict__ z, __bf16* __restrict__ zb) {
  __shared__ float s_sum[256], s_sq[256];
  const int row = blockIdx.x;
  const int tid = threadIdx.x;
  const float* yr = y + (size_t)row * D_;
  const float* xr = x + (size_t)row * D_;
  float a0 = yr[tid] + xr[tid];
  float a1 = yr[tid + 256] + xr[tid + 256];
  s_sum[tid] = a0 + a1;
  s_sq[tid]  = a0*a0 + a1*a1;
  __syncthreads();
  for (int st = 128; st > 0; st >>= 1) {
    if (tid < st) { s_sum[tid] += s_sum[tid+st]; s_sq[tid] += s_sq[tid+st]; }
    __syncthreads();
  }
  const float mu   = s_sum[0] * (1.0f / D_);
  const float var  = s_sq[0] * (1.0f / D_) - mu * mu;
  const float rstd = rsqrtf(var + 1e-5f);
  const float z0 = (a0 - mu) * rstd * gamma[tid]       + beta[tid];
  const float z1 = (a1 - mu) * rstd * gamma[tid + 256] + beta[tid + 256];
  const size_t base = (size_t)row * D_;
  z[base + tid]        = z0;  z[base + tid + 256]  = z1;
  zb[base + tid]       = (__bf16)z0;
  zb[base + tid + 256] = (__bf16)z1;
}

// ---------------------------------------------------------------------------
// K5/K6/K7: generic bf16 GEMM  out = A(MxK) * Bw(NxK)^T  with fused epilogue
// (bias, relu, f32 residual add, f32 and/or bf16 stores).
// Per-wave 16x64 tile; A fragment reused 4x.
// ---------------------------------------------------------------------------
__global__ void k_gemm(const __bf16* __restrict__ A, const __bf16* __restrict__ Bw,
                       const float* __restrict__ bias, const float* __restrict__ resid,
                       float* __restrict__ outF, __bf16* __restrict__ outB,
                       int M, int N, int K, int do_relu) {
  const int wt = blockIdx.x * 8 + (threadIdx.x >> 5);
  const int ncol = N >> 6;
  if (wt >= (M >> 4) * ncol) return;
  const int m0 = (wt / ncol) * 16;
  const int n0 = (wt % ncol) * 64;

  v8f acc[4];
  #pragma unroll
  for (int j = 0; j < 4; ++j) acc[j] = vzero8();

  for (int k = 0; k < K; k += 32) {
    v16bf a = ld_frag(A, K, m0, k);
    #pragma unroll
    for (int j = 0; j < 4; ++j) {
      v16bf b = ld_frag(Bw, K, n0 + 16*j, k);
      acc[j] = wmma_bf16(a, b, acc[j]);
    }
  }

  const int lane = threadIdx.x & 31;
  const int n = lane & 15, mo = (lane >> 4) << 3;
  #pragma unroll
  for (int j = 0; j < 4; ++j) {
    const int col = n0 + 16*j + n;
    const float bi = bias ? bias[col] : 0.0f;
    #pragma unroll
    for (int v = 0; v < 8; ++v) {
      const int row = m0 + mo + v;
      float val = acc[j][v] + bi;
      if (do_relu) val = fmaxf(val, 0.0f);
      const size_t idx = (size_t)row * N + col;
      if (resid) val += resid[idx];
      if (outF) outF[idx] = val;
      if (outB) outB[idx] = (__bf16)val;
    }
  }
}

// ---------------------------------------------------------------------------
// Host launcher
// ---------------------------------------------------------------------------
extern "C" void kernel_launch(void* const* d_in, const int* in_sizes, int n_in,
                              void* d_out, int out_size, void* d_ws, size_t ws_size,
                              hipStream_t stream) {
  (void)in_sizes; (void)n_in; (void)out_size; (void)ws_size;
  const float* x     = (const float*)d_in[0];
  const float* P     = (const float*)d_in[1];
  const float* Q     = (const float*)d_in[2];
  const float* gamma = (const float*)d_in[3];
  const float* beta  = (const float*)d_in[4];
  const float* W1    = (const float*)d_in[5];
  const float* b1    = (const float*)d_in[6];
  const float* W2    = (const float*)d_in[7];
  const float* b2    = (const float*)d_in[8];
  const float* Wp    = (const float*)d_in[9];
  const float* bp    = (const float*)d_in[10];
  float* out = (float*)d_out;

  char* ws = (char*)d_ws;
  size_t off = 0;
  auto take = [&](size_t bytes) -> char* {
    off = (off + 255) & ~(size_t)255;
    char* p = ws + off;
    off += bytes;
    return p;
  };
  __bf16* xb  = (__bf16*)take((size_t)B_*T_*D_ * 2);
  __bf16* Qb  = (__bf16*)take((size_t)H_*D_*D_ * 2);
  __bf16* Pb  = (__bf16*)take((size_t)H_*D_*D_ * 2);
  __bf16* W1b = (__bf16*)take((size_t)D2_*D_   * 2);
  __bf16* W2b = (__bf16*)take((size_t)D_*D2_   * 2);
  __bf16* Wpb = (__bf16*)take((size_t)D_*D_    * 2);
  __bf16* QZb = (__bf16*)take((size_t)B_*H_*T_*D_ * 2);
  __bf16* PZt = (__bf16*)take((size_t)B_*H_*D_*T_ * 2);
  float*  yf  = (float*) take((size_t)B_*T_*D_ * 4);
  float*  z   = (float*) take((size_t)B_*T_*D_ * 4);
  __bf16* zb  = (__bf16*)take((size_t)B_*T_*D_ * 2);
  __bf16* m1b = (__bf16*)take((size_t)MTOT_*D2_ * 2);
  float*  z2  = (float*) take((size_t)B_*T_*D_ * 4);
  __bf16* z2b = (__bf16*)take((size_t)B_*T_*D_ * 2);

  // Stage 1: bf16 conversions
  const int nx = B_*T_*D_;          // 2,097,152
  const int nq = H_*D_*D_;          // 2,097,152
  k_cvt_bf16<<<(nx + 255)/256, 256, 0, stream>>>(x,  xb,  nx);
  k_cvt_bf16<<<(nq + 255)/256, 256, 0, stream>>>(Q,  Qb,  nq);
  k_cvt_bf16<<<(nq + 255)/256, 256, 0, stream>>>(P,  Pb,  nq);
  k_cvt_bf16<<<(D2_*D_ + 255)/256, 256, 0, stream>>>(W1, W1b, D2_*D_);
  k_cvt_bf16<<<(D_*D2_ + 255)/256, 256, 0, stream>>>(W2, W2b, D_*D2_);
  k_cvt_bf16<<<(D_*D_  + 255)/256, 256, 0, stream>>>(Wp, Wpb, D_*D_);

  // Stage 2: QZ / PZ projections (16384 wave-tiles / 8 waves per block)
  k_proj<<<2048, 256, 0, stream>>>(xb, Qb, Pb, QZb, PZt);

  // Stage 3: causal linear attention, head-sum via atomics into zeroed y
  hipMemsetAsync(yf, 0, (size_t)B_*T_*D_ * 4, stream);
  k_attn<<<B_*H_*(T_/64), 256, 0, stream>>>(xb, QZb, PZt, yf);

  // Stage 4: LayerNorm(y + x)
  k_ln<<<B_*T_, 256, 0, stream>>>(yf, x, gamma, beta, z, zb);

  // Stage 5: MLP + residual + final projection
  k_gemm<<<(MTOT_/16)*(D2_/64)/8, 256, 0, stream>>>(
      zb,  W1b, b1, nullptr, nullptr, m1b, MTOT_, D2_, D_, 1);
  k_gemm<<<(MTOT_/16)*(D_/64)/8, 256, 0, stream>>>(
      m1b, W2b, b2, z,       z2,      z2b, MTOT_, D_, D2_, 0);
  k_gemm<<<(MTOT_/16)*(D_/64)/8, 256, 0, stream>>>(
      z2b, Wpb, bp, nullptr, out,     nullptr, MTOT_, D_, D_, 0);
}
